// MdTv2_66108136620628
// MI455X (gfx1250) — compile-verified, matrix operand order checked by reading, not run on hardware
//
#include <hip/hip_runtime.h>
#include <hip/hip_bf16.h>

typedef __attribute__((ext_vector_type(16))) _Float16 v16h;
typedef __attribute__((ext_vector_type(8)))  _Float16 v8h;
typedef __attribute__((ext_vector_type(2)))  _Float16 v2h;
typedef __attribute__((ext_vector_type(8)))  float    v8f;

#define NPOS   32768   // 32*32*32
#define CIN    64
#define COUT   1024
#define NH     16
#define HD     64
#define NNB    27

// ---------------------------------------------------------------------------
// Kernel 1: per-position LayerNorm over 64 channels, f32 [64, NPOS] -> f16 [NPOS, 64]
// ---------------------------------------------------------------------------
__global__ __launch_bounds__(256) void ln_kernel(const float* __restrict__ X,
                                                 const float* __restrict__ gamma,
                                                 const float* __restrict__ beta,
                                                 _Float16* __restrict__ out) {
    int pos = blockIdx.x * 256 + threadIdx.x;
    if (pos >= NPOS) return;
    float s = 0.f, s2 = 0.f;
#pragma unroll 8
    for (int c = 0; c < CIN; ++c) {
        float v = X[(size_t)c * NPOS + pos];
        s += v; s2 += v * v;
    }
    float mu  = s * (1.0f / CIN);
    float var = s2 * (1.0f / CIN) - mu * mu;
    float inv = rsqrtf(var + 1e-5f);
#pragma unroll 8
    for (int c = 0; c < CIN; ++c) {
        float v = X[(size_t)c * NPOS + pos];
        out[(size_t)pos * CIN + c] = (_Float16)(((v - mu) * inv) * gamma[c] + beta[c]);
    }
}

// ---------------------------------------------------------------------------
// Kernel 2: repack weights f32 [64,1024] into per-lane WMMA B-fragment order.
//   lane 0-15:  N = tileN*16+lane,    halves 0..7 -> K kb*32+0..7,  8..15 -> K kb*32+16..23
//   lane 16-31: N = tileN*16+lane-16, halves 0..7 -> K kb*32+8..15, 8..15 -> K kb*32+24..31
// Packed index: wp[((tileN*2 + kb)*32 + lane)*16 + i]
// ---------------------------------------------------------------------------
__global__ __launch_bounds__(256) void repack_w_kernel(const float* __restrict__ w,
                                                       _Float16* __restrict__ wp) {
    int idx = blockIdx.x * 256 + threadIdx.x;          // 0 .. 65535
    int i     = idx & 15;
    int lane  = (idx >> 4) & 31;
    int kb    = (idx >> 9) & 1;
    int tileN = idx >> 10;                             // 0..63
    int n     = tileN * 16 + (lane & 15);
    int khalf = (lane >> 4) * 8;
    int k     = kb * 32 + khalf + (i < 8 ? i : i + 8);
    wp[idx] = (_Float16)w[(size_t)k * COUT + n];
}

// ---------------------------------------------------------------------------
// Kernel 3: GEMM [NPOS,64](f16) @ [64,1024](f16, packed frags) + bias -> [NPOS,1024](f16)
// One wave per 16(M) x 64(N) strip: 4 N-tiles, K=64 => 2 WMMAs per tile (8 total).
// ---------------------------------------------------------------------------
__global__ __launch_bounds__(256) void gemm_wmma(const _Float16* __restrict__ A,    // [NPOS, 64]
                                                 const _Float16* __restrict__ Wp,   // packed frags
                                                 const float* __restrict__ bias,    // [1024]
                                                 _Float16* __restrict__ Out) {      // [NPOS, 1024]
    const int lane   = threadIdx.x & 31;
    const int wave   = threadIdx.x >> 5;
    const int job    = blockIdx.x * 8 + wave;          // 32768 jobs
    const int tileM  = job >> 4;                       // 0..2047
    const int nGroup = job & 15;                       // 0..15 (group of 4 N-tiles)

    const int mrow  = tileM * 16 + (lane & 15);
    const int khalf = (lane >> 4) * 8;                 // 0 or 8

    const _Float16* arow = A + (size_t)mrow * CIN;
    v8h a0lo = *(const v8h*)(arow + khalf);
    v8h a0hi = *(const v8h*)(arow + khalf + 16);
    v8h a1lo = *(const v8h*)(arow + khalf + 32);
    v8h a1hi = *(const v8h*)(arow + khalf + 48);
    v16h a0, a1;
#pragma unroll
    for (int i = 0; i < 8; ++i) { a0[i] = a0lo[i]; a0[8 + i] = a0hi[i];
                                  a1[i] = a1lo[i]; a1[8 + i] = a1hi[i]; }

    const v16h* WP = (const v16h*)Wp;                  // 32B-aligned fragment rows

#pragma unroll
    for (int nt = 0; nt < 4; ++nt) {
        const int tileN = nGroup * 4 + nt;
        const int ncol  = tileN * 16 + (lane & 15);

        v16h b0 = WP[(size_t)(tileN * 2 + 0) * 32 + lane];
        v16h b1 = WP[(size_t)(tileN * 2 + 1) * 32 + lane];

        v8f c = {};
        c = __builtin_amdgcn_wmma_f32_16x16x32_f16(false, a0, false, b0,
                                                   (short)0, c, false, false);
        c = __builtin_amdgcn_wmma_f32_16x16x32_f16(false, a1, false, b1,
                                                   (short)0, c, false, false);

        const float bv = bias[ncol];
#pragma unroll
        for (int i = 0; i < 8; ++i) {
            const int m = tileM * 16 + i + 8 * (lane >> 4);
            Out[(size_t)m * COUT + ncol] = (_Float16)(c[i] + bv);
        }
    }
}

// ---------------------------------------------------------------------------
// Kernel 4: neighborhood attention. Thread = (pos, head).
// 64-dim dots accumulated in packed f16 pairs -> v_pk_fma_f16 (2 MACs/instr).
// v2h extraction via even-offset shufflevector is register-exact (free).
// ---------------------------------------------------------------------------
__global__ __launch_bounds__(256) void nattn_kernel(const _Float16* __restrict__ Q,
                                                    const _Float16* __restrict__ K,
                                                    const float* __restrict__ rpb,
                                                    float* __restrict__ out) {
    const int tid  = blockIdx.x * 256 + threadIdx.x;   // 0 .. NPOS*NH-1
    const int head = tid & 15;
    const int pos  = tid >> 4;
    const int t = pos & 31, w = (pos >> 5) & 31, h = pos >> 10;

    const _Float16* qrow = Q + (size_t)pos * COUT + head * HD;
    v8h qv[8];
#pragma unroll
    for (int i = 0; i < 8; ++i) qv[i] = *(const v8h*)(qrow + i * 8);

    float sc[NNB];
#pragma unroll
    for (int dz = 0; dz < 3; ++dz)
#pragma unroll
    for (int dy = 0; dy < 3; ++dy)
#pragma unroll
    for (int dx = 0; dx < 3; ++dx) {
        const int nb = (dz * 3 + dy) * 3 + dx;
        const int hh = h + dz - 1, ww = w + dy - 1, tt = t + dx - 1;
        float acc = 0.f;
        if ((unsigned)hh < 32u && (unsigned)ww < 32u && (unsigned)tt < 32u) {
            const int p2 = (hh * 32 + ww) * 32 + tt;
            const _Float16* krow = K + (size_t)p2 * COUT + head * HD;
            v2h acc0 = {(_Float16)0.f, (_Float16)0.f};
            v2h acc1 = {(_Float16)0.f, (_Float16)0.f};
            v2h acc2 = {(_Float16)0.f, (_Float16)0.f};
            v2h acc3 = {(_Float16)0.f, (_Float16)0.f};
#pragma unroll
            for (int i = 0; i < 8; ++i) {
                v8h kv = *(const v8h*)(krow + i * 8);
                v2h q0 = __builtin_shufflevector(qv[i], qv[i], 0, 1);
                v2h q1 = __builtin_shufflevector(qv[i], qv[i], 2, 3);
                v2h q2 = __builtin_shufflevector(qv[i], qv[i], 4, 5);
                v2h q3 = __builtin_shufflevector(qv[i], qv[i], 6, 7);
                v2h k0 = __builtin_shufflevector(kv, kv, 0, 1);
                v2h k1 = __builtin_shufflevector(kv, kv, 2, 3);
                v2h k2 = __builtin_shufflevector(kv, kv, 4, 5);
                v2h k3 = __builtin_shufflevector(kv, kv, 6, 7);
                acc0 += q0 * k0;     // v_pk_fma_f16
                acc1 += q1 * k1;
                acc2 += q2 * k2;
                acc3 += q3 * k3;
            }
            v2h accp = (acc0 + acc1) + (acc2 + acc3);
            acc = (float)accp[0] + (float)accp[1];
        }
        sc[nb] = acc + rpb[head * NNB + nb];
    }

    // softmax over 27
    float mx = sc[0];
#pragma unroll
    for (int i = 1; i < NNB; ++i) mx = fmaxf(mx, sc[i]);
    float ssum = 0.f;
#pragma unroll
    for (int i = 0; i < NNB; ++i) { float e = __expf(sc[i] - mx); sc[i] = e; ssum += e; }
    const float inv = 1.0f / ssum;

    float x0 = 0.f, x1 = 0.f, x2 = 0.f;
#pragma unroll
    for (int dz = 0; dz < 3; ++dz)
#pragma unroll
    for (int dy = 0; dy < 3; ++dy)
#pragma unroll
    for (int dx = 0; dx < 3; ++dx) {
        const int nb = (dz * 3 + dy) * 3 + dx;
        const float a = sc[nb] * inv;
        x0 += a * (float)(dz - 1);
        x1 += a * (float)(dy - 1);
        x2 += a * (float)(dx - 1);
    }

    out[(size_t)(head * 3 + 0) * NPOS + pos] = x0;
    out[(size_t)(head * 3 + 1) * NPOS + pos] = x1;
    out[(size_t)(head * 3 + 2) * NPOS + pos] = x2;
}

// ---------------------------------------------------------------------------
// Host launcher
// ---------------------------------------------------------------------------
extern "C" void kernel_launch(void* const* d_in, const int* in_sizes, int n_in,
                              void* d_out, int out_size, void* d_ws, size_t ws_size,
                              hipStream_t stream) {
    (void)in_sizes; (void)n_in; (void)out_size; (void)ws_size;

    const float* F       = (const float*)d_in[0];
    const float* M       = (const float*)d_in[1];
    const float* gamma_f = (const float*)d_in[2];
    const float* beta_f  = (const float*)d_in[3];
    const float* w_f     = (const float*)d_in[4];
    const float* b_f     = (const float*)d_in[5];
    const float* gamma_m = (const float*)d_in[6];
    const float* beta_m  = (const float*)d_in[7];
    const float* w_m     = (const float*)d_in[8];
    const float* b_m     = (const float*)d_in[9];
    const float* rpb     = (const float*)d_in[10];
    float* out           = (float*)d_out;

    // workspace layout (all 256B aligned)
    char* ws = (char*)d_ws;
    size_t off = 0;
    auto alloc = [&](size_t bytes) {
        char* p = ws + off;
        off += (bytes + 255) & ~(size_t)255;
        return p;
    };
    _Float16* lnF  = (_Float16*)alloc((size_t)NPOS * CIN * 2);    //  4 MB
    _Float16* lnM  = (_Float16*)alloc((size_t)NPOS * CIN * 2);    //  4 MB
    _Float16* wf16 = (_Float16*)alloc((size_t)CIN * COUT * 2);    // 128 KB packed frags
    _Float16* wm16 = (_Float16*)alloc((size_t)CIN * COUT * 2);    // 128 KB packed frags
    _Float16* q16  = (_Float16*)alloc((size_t)NPOS * COUT * 2);   // 64 MB
    _Float16* k16  = (_Float16*)alloc((size_t)NPOS * COUT * 2);   // 64 MB

    // 1) LayerNorm both inputs
    ln_kernel<<<NPOS / 256, 256, 0, stream>>>(F, gamma_f, beta_f, lnF);
    ln_kernel<<<NPOS / 256, 256, 0, stream>>>(M, gamma_m, beta_m, lnM);

    // 2) Weight repack into WMMA B-fragment order (f16)
    repack_w_kernel<<<(CIN * COUT) / 256, 256, 0, stream>>>(w_f, wf16);
    repack_w_kernel<<<(CIN * COUT) / 256, 256, 0, stream>>>(w_m, wm16);

    // 3) WMMA projection GEMMs: q = lnF @ w_f + b_f ; k = lnM @ w_m + b_m
    gemm_wmma<<<4096, 256, 0, stream>>>(lnF, wf16, b_f, q16);
    gemm_wmma<<<4096, 256, 0, stream>>>(lnM, wm16, b_m, k16);

    // 4) Neighborhood attention + softmax + V_GRID contraction
    nattn_kernel<<<(NPOS * NH) / 256, 256, 0, stream>>>(q16, k16, rpb, out);
}
